// XMemBackboneWrapper_6562710028908
// MI455X (gfx1250) — matrix operand order repaired and checked by reading, not compiled, not run on hardware
//
#include <hip/hip_runtime.h>
#include <hip/hip_bf16.h>
#include <math.h>

// ---------------- static problem config (mirrors reference) ----------------
#define BATCH 2
#define TT    16
#define HH    512
#define WWID  512
#define HWN   1024          // 32x32 patches
#define CK    64
#define CF    256
#define CV    512
#define CH    64
#define KPATCH 768          // 3*16*16
#define NPROJ  448          // 385 padded to 64-col macro tiles
#define NMAXM  3072         // max memory entries (3 frames * 1024)
#define KQ     128          // [-mk^2 ; mk] similarity K
#define KVAL   288          // 257 padded to 32
#define KH2    576
#define KHD    832
#define NDEC   64           // 1 padded to 64
#define SMITERS (NMAXM/256)
#define MT 128              // GEMM macro-tile rows
#define NT 64               // GEMM macro-tile cols

typedef __attribute__((ext_vector_type(16))) __bf16 v16bf;
typedef __attribute__((ext_vector_type(8)))  float  v8f;

__device__ __forceinline__ unsigned short f2bf(float f) {
  unsigned int u = __float_as_uint(f);
  unsigned int r = u + 0x7FFFu + ((u >> 16) & 1u);   // round-to-nearest-even
  return (unsigned short)(r >> 16);
}

__device__ __forceinline__ unsigned lds_off(const void* p) {
  // LDS aperture generic pointers carry the LDS byte offset in bits [31:0]
  return (unsigned)(unsigned long long)p;
}

__device__ __forceinline__ void async_b128(unsigned lds, const void* g) {
  unsigned long long ga = (unsigned long long)g;
  asm volatile("global_load_async_to_lds_b128 %0, %1, off"
               :: "v"(lds), "v"(ga) : "memory");
}

// ---------------------------------------------------------------------------
// bf16 WMMA GEMM, LDS double-buffered with async global->LDS staging.
//   C[M,N](f32) = A[M,K](bf16 row-major, lda) x Wt[N,K](bf16 N-major, ldw)
// Block = 256 thr (8 waves) computes a 128x64 macro-tile. Wave w owns rows
// [w*16, w*16+16) and all four 16x16 N-tiles: 1 A fragment feeds 4 WMMAs.
// All fragment ds_loads are issued as one clause before the 4 WMMAs so a
// single s_wait_dscnt covers them and the WMMAs issue back-to-back.
// Requires M%128==0, N%64==0, K%32==0 (guaranteed by padding).
// ---------------------------------------------------------------------------
__global__ __launch_bounds__(256)
void gemm_bf16_wmma(const unsigned short* __restrict__ A, int lda, long sA,
                    const unsigned short* __restrict__ Wt, int ldw, long sW,
                    float* __restrict__ C, int ldc, long sC,
                    int M, int K, int N) {
  __shared__ uint4 Ash[2][MT * 4];   // 128 rows x 32 bf16 (64B/row), x2 bufs
  __shared__ uint4 Bsh[2][NT * 4];   // 64 rows x 32 bf16

  const int b = blockIdx.y;
  A  += (long)b * sA;
  Wt += (long)b * sW;
  C  += (long)b * sC;

  const int tilesN = N / NT;
  const int tm = blockIdx.x / tilesN;
  const int tn = blockIdx.x % tilesN;
  const int tid  = threadIdx.x;
  const int wave = tid >> 5;
  const int lane = tid & 31;
  const int half = lane >> 4;
  const int l15  = lane & 15;

  // per-thread async chunk sources: A -> 2 x 16B chunks, B -> 1 x 16B chunk
  const int ac0 = tid * 2, ac1 = tid * 2 + 1;             // A chunk ids 0..511
  const unsigned short* aT0 = A + (long)(tm * MT + (ac0 >> 2)) * lda + (ac0 & 3) * 8;
  const unsigned short* aT1 = A + (long)(tm * MT + (ac1 >> 2)) * lda + (ac1 & 3) * 8;
  const unsigned short* bT  = Wt + (long)(tn * NT + (tid >> 2)) * ldw + (tid & 3) * 8;

  auto stage = [&](int kb, int buf) {
    async_b128(lds_off(&Ash[buf][ac0]), aT0 + kb);
    async_b128(lds_off(&Ash[buf][ac1]), aT1 + kb);
    async_b128(lds_off(&Bsh[buf][tid]), bT + kb);
  };

  v8f acc[4];
  #pragma unroll
  for (int nt = 0; nt < 4; ++nt) acc[nt] = (v8f){0.f,0.f,0.f,0.f,0.f,0.f,0.f,0.f};

  const int S = K >> 5;
  stage(0, 0);
  for (int s = 0; s < S; ++s) {
    const int buf = s & 1;
    if (s + 1 < S) {
      stage((s + 1) << 5, buf ^ 1);                 // overlap next stage
      asm volatile("s_wait_asynccnt 0x3" ::: "memory");
    } else {
      asm volatile("s_wait_asynccnt 0x0" ::: "memory");
    }
    __syncthreads();

    // Preload A fragment and ALL four B fragments (one ds clause, one wait),
    // then issue the 4 WMMAs back-to-back (independent accumulators).
    union { v16bf v; uint4 q[2]; } af, bfr[4];
    const int arow = wave * 16 + l15;
    af.q[0] = Ash[buf][arow * 4 + half];
    af.q[1] = Ash[buf][arow * 4 + 2 + half];
    #pragma unroll
    for (int nt = 0; nt < 4; ++nt) {
      const int brow = nt * 16 + l15;
      bfr[nt].q[0] = Bsh[buf][brow * 4 + half];
      bfr[nt].q[1] = Bsh[buf][brow * 4 + 2 + half];
    }
    #pragma unroll
    for (int nt = 0; nt < 4; ++nt)
      acc[nt] = __builtin_amdgcn_wmma_f32_16x16x32_bf16(
          false, af.v, false, bfr[nt].v, (short)0, acc[nt], false, false);
    __syncthreads();
  }

  // C/D layout: VGPR i -> row base + i (+8 for upper lane half), col = lane%16
  const int rbase = tm * MT + wave * 16 + half * 8;
  #pragma unroll
  for (int nt = 0; nt < 4; ++nt) {
    const int col = tn * NT + nt * 16 + l15;
    #pragma unroll
    for (int i = 0; i < 8; ++i)
      C[(long)(rbase + i) * ldc + col] = acc[nt][i];
  }
}

// ---------------------------- packing kernels ------------------------------
__global__ void zero_f32(float* p, long n) {
  long i = (long)blockIdx.x * 256 + threadIdx.x;
  if (i < n) p[i] = 0.f;
}

// combined projection weights, N-major: cols [key(64) | shr(1) | sel(64) | f16(256) | pad]
__global__ void pack_proj(unsigned short* dst, const float* Wk, const float* Wsh,
                          const float* Wse, const float* Wf) {
  int idx = blockIdx.x * 256 + threadIdx.x;
  if (idx >= NPROJ * KPATCH) return;
  int n = idx / KPATCH, k = idx % KPATCH;
  float v = 0.f;
  if (n < CK)        v = Wk[k * CK + n];
  else if (n == CK)  v = Wsh[k];
  else if (n < 129)  v = Wse[k * CK + (n - 65)];
  else if (n < 385)  v = Wf[k * CF + (n - 129)];
  dst[idx] = f2bf(v);
}

// generic transpose-pack: dst[n*Kp+k] = src[k*Nsrc+n] with zero padding
__global__ void pack_wt(unsigned short* dst, const float* src,
                        int Nout, int Kp, int Ksrc, int Nsrc) {
  int idx = blockIdx.x * 256 + threadIdx.x;
  if (idx >= Nout * Kp) return;
  int n = idx / Kp, k = idx % Kp;
  float v = (k < Ksrc && n < Nsrc) ? src[k * Nsrc + n] : 0.f;
  dst[idx] = f2bf(v);
}

// frames[b,t,3,512,512] f32 -> patches[b, p(1024), e(768)] bf16 for frame t
__global__ void patchify(const float* frames, unsigned short* patches, int t) {
  long idx = (long)blockIdx.x * 256 + threadIdx.x;
  if (idx >= (long)BATCH * HWN * KPATCH) return;
  int e = (int)(idx % KPATCH);
  int p = (int)((idx / KPATCH) % HWN);
  int b = (int)(idx / ((long)KPATCH * HWN));
  int c = e >> 8, w = e & 255, pi = w >> 4, pj = w & 15;
  int i = p >> 5, j = p & 31;
  long src = ((long)(b * TT + t) * 3 + c);
  src = (src * HH + (i * 16 + pi)) * WWID + (j * 16 + pj);
  patches[idx] = f2bf(frames[src]);
}

// ---------------------------- per-frame kernels ----------------------------
// feat(t=0) = keys.mean over spatial
__global__ void feat0_kernel(const float* Cproj, float* out, int t) {
  __shared__ float sh[256];
  int b = blockIdx.y, c = blockIdx.x, tid = threadIdx.x;
  float s = 0.f;
  for (int p = tid; p < HWN; p += 256)
    s += Cproj[((long)b * HWN + p) * NPROJ + c];
  sh[tid] = s; __syncthreads();
  for (int st = 128; st > 0; st >>= 1) {
    if (tid < st) sh[tid] += sh[tid + st];
    __syncthreads();
  }
  if (tid == 0) out[((long)b * TT + t) * CH + c] = sh[0] * (1.f / HWN);
}

// qB[p, 0:64]=qe(sigmoid sel), qB[p,64:128]=2*qk*qe ; bsq[p]=sum qe*qk^2
__global__ void build_query(const float* Cproj, unsigned short* qB, float* bsq) {
  int idx = blockIdx.x * 256 + threadIdx.x;
  if (idx >= BATCH * HWN) return;
  int b = idx / HWN, p = idx % HWN;
  const float* row = Cproj + ((long)b * HWN + p) * NPROJ;
  unsigned short* q = qB + ((long)b * HWN + p) * KQ;
  float acc = 0.f;
  for (int c = 0; c < CK; ++c) {
    float k = row[c];
    float qe = 1.f / (1.f + __expf(-row[65 + c]));
    q[c]      = f2bf(qe);
    q[64 + c] = f2bf(2.f * k * qe);
    acc += qe * k * k;
  }
  bsq[b * HWN + p] = acc;
}

// column-wise softmax over memory entries; writes aff^T [p][n] bf16
__global__ __launch_bounds__(256)
void softmax_aff(const float* simraw, const float* bsq, const float* mem_s,
                 unsigned short* aff_t, int N) {
  __shared__ float sh[256];
  int b = blockIdx.y, p = blockIdx.x, tid = threadIdx.x;
  const float* srow = simraw + (long)b * NMAXM * HWN;
  const float bq = bsq[b * HWN + p];
  float vals[SMITERS];
  float mx = -1e30f;
  int iters = (N + 255) >> 8;
  for (int i = 0; i < iters; ++i) {
    int n = (i << 8) + tid;
    float v = -1e30f;
    if (n < N) v = (srow[(long)n * HWN + p] - bq) * mem_s[b * NMAXM + n] * 0.125f;
    vals[i] = v;
    mx = fmaxf(mx, v);
  }
  sh[tid] = mx; __syncthreads();
  for (int s = 128; s > 0; s >>= 1) {
    if (tid < s) sh[tid] = fmaxf(sh[tid], sh[tid + s]);
    __syncthreads();
  }
  mx = sh[0]; __syncthreads();
  float sum = 0.f;
  for (int i = 0; i < iters; ++i) {
    int n = (i << 8) + tid;
    float e = (n < N) ? __expf(vals[i] - mx) : 0.f;
    vals[i] = e; sum += e;
  }
  sh[tid] = sum; __syncthreads();
  for (int s = 128; s > 0; s >>= 1) {
    if (tid < s) sh[tid] += sh[tid + s];
    __syncthreads();
  }
  float inv = 1.f / sh[0];
  unsigned short* arow = aff_t + ((long)b * HWN + p) * NMAXM;
  for (int i = 0; i < iters; ++i) {
    int n = (i << 8) + tid;
    if (n < N) arow[n] = f2bf(vals[i] * inv);
  }
}

// X1 = [f16 | readout | hidden] bf16
__global__ void assembleX1(const float* Cproj, const float* readout,
                           const float* hidden, unsigned short* X1) {
  long idx = (long)blockIdx.x * 256 + threadIdx.x;
  if (idx >= (long)BATCH * HWN * KHD) return;
  int col = (int)(idx % KHD);
  long bp = idx / KHD;                  // b*HWN + p
  float v;
  if (col < 256)      v = Cproj[bp * NPROJ + 129 + col];
  else if (col < 768) v = readout[bp * CV + (col - 256)];
  else                v = hidden[bp * CH + (col - 768)];
  X1[idx] = f2bf(v);
}

// tanh epilogue of decoder-hidden GEMM; patches X1's hidden slot (-> X2)
__global__ void tanh_hd(const float* hdraw, float* hlocal, unsigned short* X1,
                        float* hidden, int update) {
  long idx = (long)blockIdx.x * 256 + threadIdx.x;
  if (idx >= (long)BATCH * HWN * CH) return;
  int c = (int)(idx % CH);
  long bp = idx / CH;
  float hl = tanhf(hdraw[idx]);
  hlocal[idx] = hl;
  X1[bp * KHD + 768 + c] = f2bf(hl);
  if (update) hidden[idx] = hl;
}

// feat[t] = sum_p sigmoid(fg)*hlocal / max(sum sigmoid(fg), eps)
__global__ void feat_kernel(const float* fgbuf, const float* hlocal,
                            float* out, int t) {
  __shared__ float sh[256];
  int b = blockIdx.y, c = blockIdx.x, tid = threadIdx.x;
  float sm = 0.f, smh = 0.f;
  for (int p = tid; p < HWN; p += 256) {
    float m = 1.f / (1.f + __expf(-fgbuf[((long)b * HWN + p) * NDEC]));
    sm  += m;
    smh += m * hlocal[((long)b * HWN + p) * CH + c];
  }
  sh[tid] = smh; __syncthreads();
  for (int s = 128; s > 0; s >>= 1) { if (tid < s) sh[tid] += sh[tid + s]; __syncthreads(); }
  float tot_mh = sh[0]; __syncthreads();
  sh[tid] = sm; __syncthreads();
  for (int s = 128; s > 0; s >>= 1) { if (tid < s) sh[tid] += sh[tid + s]; __syncthreads(); }
  if (tid == 0)
    out[((long)b * TT + t) * CH + c] = tot_mh / fmaxf(sh[0], 1e-6f);
}

// ---------------------------- memory-frame kernels -------------------------
// wmask for t=0: aggregate(init_mask) fg prob, 16x16 average-pooled
__global__ void wmask0_kernel(const float* init_masks, float* wm) {
  int idx = blockIdx.x * 256 + threadIdx.x;
  if (idx >= BATCH * HWN) return;
  int b = idx / HWN, p = idx % HWN;
  int i = p >> 5, j = p & 31;
  float acc = 0.f;
  for (int pi = 0; pi < 16; ++pi)
    for (int pj = 0; pj < 16; ++pj) {
      float v = init_masks[(long)b * HH * WWID + (i * 16 + pi) * WWID + (j * 16 + pj)];
      float bg = 1.f - v;
      float vc  = fminf(fmaxf(v,  1e-7f), 1.f - 1e-7f);
      float bgc = fminf(fmaxf(bg, 1e-7f), 1.f - 1e-7f);
      float l0 = __logf(bgc / (1.f - bgc));
      float l1 = __logf(vc  / (1.f - vc));
      acc += 1.f / (1.f + __expf(l0 - l1));   // softmax fg of [l0,l1]
    }
  wm[idx] = acc * (1.f / 256.f);
}

// X3 = [f16 | mask | zero-pad] bf16; mask from wm (t==0) or sigmoid(fg)
__global__ void assembleX3(const float* Cproj, const float* wm,
                           const float* fgbuf, unsigned short* X3, int useFg) {
  long idx = (long)blockIdx.x * 256 + threadIdx.x;
  if (idx >= (long)BATCH * HWN * KVAL) return;
  int col = (int)(idx % KVAL);
  long bp = idx / KVAL;
  float v = 0.f;
  if (col < 256) v = Cproj[bp * NPROJ + 129 + col];
  else if (col == 256)
    v = useFg ? 1.f / (1.f + __expf(-fgbuf[bp * NDEC])) : wm[bp];
  X3[idx] = f2bf(v);
}

// scatter value into bf16 memory bank (CV-major) and into X4 low cols
__global__ void value_post(const float* value, unsigned short* mvb,
                           unsigned short* X4, int off) {
  long idx = (long)blockIdx.x * 256 + threadIdx.x;
  if (idx >= (long)BATCH * HWN * CV) return;
  int v = (int)(idx % CV);
  int p = (int)((idx / CV) % HWN);
  int b = (int)(idx / ((long)CV * HWN));
  unsigned short x = f2bf(value[idx]);
  mvb[((long)b * CV + v) * NMAXM + off + p] = x;
  X4[((long)b * HWN + p) * KH2 + v] = x;
}

// append [-k^2 ; k] rows and shrinkage (d^2+1) to the memory bank
__global__ void append_key(const float* Cproj, unsigned short* memA,
                           float* mem_s, int off) {
  long idx = (long)blockIdx.x * 256 + threadIdx.x;
  if (idx >= (long)BATCH * HWN * CK) return;
  int c = (int)(idx % CK);
  int p = (int)((idx / CK) % HWN);
  int b = (int)(idx / ((long)CK * HWN));
  const float* row = Cproj + ((long)b * HWN + p) * NPROJ;
  float k = row[c];
  unsigned short* dst = memA + ((long)b * NMAXM + off + p) * KQ;
  dst[c]      = f2bf(-k * k);
  dst[64 + c] = f2bf(k);
  if (c == 0) {
    float s = row[64];
    mem_s[(long)b * NMAXM + off + p] = s * s + 1.f;
  }
}

__global__ void assembleX4h(const float* hidden, unsigned short* X4) {
  long idx = (long)blockIdx.x * 256 + threadIdx.x;
  if (idx >= (long)BATCH * HWN * CH) return;
  int c = (int)(idx % CH);
  long bp = idx / CH;
  X4[bp * KH2 + 512 + c] = f2bf(hidden[idx]);
}

__global__ void tanh_h2(const float* h2raw, float* hidden) {
  long idx = (long)blockIdx.x * 256 + threadIdx.x;
  if (idx >= (long)BATCH * HWN * CH) return;
  hidden[idx] = tanhf(h2raw[idx]);
}

// ---------------------------------------------------------------------------
extern "C" void kernel_launch(void* const* d_in, const int* in_sizes, int n_in,
                              void* d_out, int out_size, void* d_ws, size_t ws_size,
                              hipStream_t stream) {
  (void)in_sizes; (void)n_in; (void)out_size; (void)ws_size;
  const float* frames     = (const float*)d_in[0];
  const float* init_masks = (const float*)d_in[1];
  /* d_in[2] init_labels unused */
  const float* W_key  = (const float*)d_in[3];
  const float* W_shr  = (const float*)d_in[4];
  const float* W_sel  = (const float*)d_in[5];
  const float* W_f16  = (const float*)d_in[6];
  const float* W_val  = (const float*)d_in[7];
  const float* W_hid2 = (const float*)d_in[8];
  const float* W_hd   = (const float*)d_in[9];
  const float* W_dec  = (const float*)d_in[10];
  float* out = (float*)d_out;

  char* ws = (char*)d_ws;
  size_t cur = 0;
  auto alloc = [&](size_t bytes) -> char* {
    char* p = ws + cur;
    cur = (cur + bytes + 255) & ~(size_t)255;
    return p;
  };

  unsigned short* WtProj = (unsigned short*)alloc((size_t)NPROJ * KPATCH * 2);
  unsigned short* WtVal  = (unsigned short*)alloc((size_t)CV * KVAL * 2);
  unsigned short* WtH2   = (unsigned short*)alloc((size_t)CH * KH2 * 2);
  unsigned short* WtHd   = (unsigned short*)alloc((size_t)CH * KHD * 2);
  unsigned short* WtDec  = (unsigned short*)alloc((size_t)NDEC * KHD * 2);
  unsigned short* patches= (unsigned short*)alloc((size_t)BATCH * HWN * KPATCH * 2);
  float* Cproj  = (float*)alloc((size_t)BATCH * HWN * NPROJ * 4);
  unsigned short* qB = (unsigned short*)alloc((size_t)BATCH * HWN * KQ * 2);
  float* bsq    = (float*)alloc((size_t)BATCH * HWN * 4);
  float* simraw = (float*)alloc((size_t)BATCH * NMAXM * HWN * 4);
  unsigned short* aff = (unsigned short*)alloc((size_t)BATCH * HWN * NMAXM * 2);
  float* readout = (float*)alloc((size_t)BATCH * HWN * CV * 4);
  unsigned short* X1 = (unsigned short*)alloc((size_t)BATCH * HWN * KHD * 2);
  float* hdraw  = (float*)alloc((size_t)BATCH * HWN * CH * 4);
  float* hlocal = (float*)alloc((size_t)BATCH * HWN * CH * 4);
  float* fgbuf  = (float*)alloc((size_t)BATCH * HWN * NDEC * 4);
  float* hidden = (float*)alloc((size_t)BATCH * HWN * CH * 4);
  unsigned short* memA = (unsigned short*)alloc((size_t)BATCH * NMAXM * KQ * 2);
  float* mem_s  = (float*)alloc((size_t)BATCH * NMAXM * 4);
  unsigned short* mvb = (unsigned short*)alloc((size_t)BATCH * CV * NMAXM * 2);
  unsigned short* X3  = (unsigned short*)alloc((size_t)BATCH * HWN * KVAL * 2);
  float* value  = (float*)alloc((size_t)BATCH * HWN * CV * 4);
  unsigned short* X4 = (unsigned short*)alloc((size_t)BATCH * HWN * KH2 * 2);
  float* h2raw  = (float*)alloc((size_t)BATCH * HWN * CH * 4);
  float* wm     = (float*)alloc((size_t)BATCH * HWN * 4);

  auto blocks = [](long n) { return dim3((unsigned)((n + 255) / 256)); };
  auto gemm = [&](const unsigned short* A, int lda, long sA,
                  const unsigned short* Wt, int ldw, long sW,
                  float* C, int ldc, long sC, int M, int K, int N) {
    dim3 grid((M / MT) * (N / NT), BATCH);
    gemm_bf16_wmma<<<grid, 256, 0, stream>>>(A, lda, sA, Wt, ldw, sW,
                                             C, ldc, sC, M, K, N);
  };

  // ---- one-time packing / init ----
  pack_proj<<<blocks((long)NPROJ * KPATCH), 256, 0, stream>>>(WtProj, W_key, W_shr, W_sel, W_f16);
  pack_wt<<<blocks((long)CV * KVAL), 256, 0, stream>>>(WtVal, W_val, CV, KVAL, 257, CV);
  pack_wt<<<blocks((long)CH * KH2), 256, 0, stream>>>(WtH2, W_hid2, CH, KH2, KH2, CH);
  pack_wt<<<blocks((long)CH * KHD), 256, 0, stream>>>(WtHd, W_hd, CH, KHD, KHD, CH);
  pack_wt<<<blocks((long)NDEC * KHD), 256, 0, stream>>>(WtDec, W_dec, NDEC, KHD, KHD, 1);
  zero_f32<<<blocks((long)BATCH * HWN * CH), 256, 0, stream>>>(hidden, (long)BATCH * HWN * CH);
  wmask0_kernel<<<blocks((long)BATCH * HWN), 256, 0, stream>>>(init_masks, wm);

  // ---- static XMem schedule: mem frames t=0,5,10; deep updates t=0,10 ----
  int nmem = 0;
  for (int t = 0; t < TT; ++t) {
    patchify<<<blocks((long)BATCH * HWN * KPATCH), 256, 0, stream>>>(frames, patches, t);
    gemm(patches, KPATCH, (long)HWN * KPATCH, WtProj, KPATCH, 0,
         Cproj, NPROJ, (long)HWN * NPROJ, HWN, KPATCH, NPROJ);

    if (t == 0) {
      feat0_kernel<<<dim3(CH, BATCH), 256, 0, stream>>>(Cproj, out, 0);
    } else {
      int N = nmem * HWN;
      build_query<<<blocks((long)BATCH * HWN), 256, 0, stream>>>(Cproj, qB, bsq);
      gemm(memA, KQ, (long)NMAXM * KQ, qB, KQ, (long)HWN * KQ,
           simraw, HWN, (long)NMAXM * HWN, N, KQ, HWN);
      softmax_aff<<<dim3(HWN, BATCH), 256, 0, stream>>>(simraw, bsq, mem_s, aff, N);
      gemm(aff, NMAXM, (long)HWN * NMAXM, mvb, NMAXM, (long)CV * NMAXM,
           readout, CV, (long)HWN * CV, HWN, N, CV);
      assembleX1<<<blocks((long)BATCH * HWN * KHD), 256, 0, stream>>>(Cproj, readout, hidden, X1);
      gemm(X1, KHD, (long)HWN * KHD, WtHd, KHD, 0,
           hdraw, CH, (long)HWN * CH, HWN, KHD, CH);
      tanh_hd<<<blocks((long)BATCH * HWN * CH), 256, 0, stream>>>(
          hdraw, hlocal, X1, hidden, (t != TT - 1) ? 1 : 0);
      gemm(X1, KHD, (long)HWN * KHD, WtDec, KHD, 0,
           fgbuf, NDEC, (long)HWN * NDEC, HWN, KHD, NDEC);
      feat_kernel<<<dim3(CH, BATCH), 256, 0, stream>>>(fgbuf, hlocal, out, t);
    }

    bool is_mem  = (t == 0 || t == 5 || t == 10);
    bool is_deep = (t == 0 || t == 10);
    if (is_mem) {
      int off = nmem * HWN;
      assembleX3<<<blocks((long)BATCH * HWN * KVAL), 256, 0, stream>>>(
          Cproj, wm, fgbuf, X3, t > 0 ? 1 : 0);
      gemm(X3, KVAL, (long)HWN * KVAL, WtVal, KVAL, 0,
           value, CV, (long)HWN * CV, HWN, KVAL, CV);
      value_post<<<blocks((long)BATCH * HWN * CV), 256, 0, stream>>>(value, mvb, X4, off);
      append_key<<<blocks((long)BATCH * HWN * CK), 256, 0, stream>>>(Cproj, memA, mem_s, off);
      nmem++;
      if (is_deep) {
        assembleX4h<<<blocks((long)BATCH * HWN * CH), 256, 0, stream>>>(hidden, X4);
        gemm(X4, KH2, (long)HWN * KH2, WtH2, KH2, 0,
             h2raw, CH, (long)HWN * CH, HWN, KH2, CH);
        tanh_h2<<<blocks((long)BATCH * HWN * CH), 256, 0, stream>>>(h2raw, hidden);
      }
    }
  }
}
